// MultiQueryAttention_73134703116474
// MI455X (gfx1250) — matrix operand order repaired and checked by reading, not compiled, not run on hardware
//
#include <hip/hip_runtime.h>
#include <hip/hip_bf16.h>

typedef __bf16 bf16_t;
typedef __attribute__((ext_vector_type(16))) __bf16 v16bf;
typedef __attribute__((ext_vector_type(8)))  __bf16 v8bf;
typedef __attribute__((ext_vector_type(8)))  float  v8f;

// ---------------------------------------------------------------------------
// WMMA wrapper: D = A(16x32 bf16) * B(32x16 bf16) + C(16x16 f32)
// ---------------------------------------------------------------------------
__device__ __forceinline__ v8f wmma_bf16(v16bf a, v16bf b, v8f c) {
  return __builtin_amdgcn_wmma_f32_16x16x32_bf16(
      /*neg_a=*/false, a, /*neg_b=*/false, b,
      /*c_mod=*/(short)0, c, /*reuse_a=*/false, /*reuse_b=*/false);
}

// ---------------------------------------------------------------------------
// Fragment loaders (contiguous k order used by both A and B, wave32)
// A 16x32 bf16: lane L -> row (L&15), K = (L>>4)*16 + {0..15}
// Works for global or LDS pointers (two 16B loads, needs 16B alignment only).
// ---------------------------------------------------------------------------
__device__ __forceinline__ v16bf load_a_bf16(const bf16_t* A, int lda, int row0, int col0) {
  int lane = threadIdx.x & 31;
  const bf16_t* p = A + (size_t)(row0 + (lane & 15)) * lda + col0 + ((lane >> 4) << 4);
  v8bf lo = *(const v8bf*)p;          // K = kb .. kb+7
  v8bf hi = *(const v8bf*)(p + 8);    // K = kb+8 .. kb+15
  v16bf a;
#pragma unroll
  for (int i = 0; i < 8; ++i) { a[i] = lo[i]; a[i + 8] = hi[i]; }
  return a;
}

__device__ __forceinline__ v16bf load_a_f32(const float* A, int lda, int row0, int col0) {
  int lane = threadIdx.x & 31;
  const float* p = A + (size_t)(row0 + (lane & 15)) * lda + col0 + ((lane >> 4) << 4);
  v16bf a;
#pragma unroll
  for (int i = 0; i < 8; ++i) {
    a[i]     = (bf16_t)p[i];
    a[i + 8] = (bf16_t)p[8 + i];
  }
  return a;
}

// B 32x16 bf16 from an N x K (row-major) matrix: lane L -> col N = (L&15),
// K = (L>>4)*16 + e.  Global variant: one contiguous 32-byte load.
__device__ __forceinline__ v16bf load_b_nk(const bf16_t* Bt, int ldb, int n0, int k0) {
  int lane = threadIdx.x & 31;
  const bf16_t* p = Bt + (size_t)(n0 + (lane & 15)) * ldb + k0 + ((lane >> 4) << 4);
  return *(const v16bf*)p;
}

// LDS variant: two 16B reads (row stride may only be 16B-aligned due to padding).
__device__ __forceinline__ v16bf load_b_lds(const bf16_t* Bt, int ldb, int n0, int k0) {
  int lane = threadIdx.x & 31;
  const bf16_t* p = Bt + (n0 + (lane & 15)) * ldb + k0 + ((lane >> 4) << 4);
  v8bf lo = *(const v8bf*)p;
  v8bf hi = *(const v8bf*)(p + 8);
  v16bf b;
#pragma unroll
  for (int i = 0; i < 8; ++i) { b[i] = lo[i]; b[i + 8] = hi[i]; }
  return b;
}

// ---------------------------------------------------------------------------
// Weight transpose + f32 -> bf16 convert via LDS tile (coalesced both sides):
//   Wt[n*K + k] = (bf16) W[k*N + n]      grid = (K/32, N/32), block = (32, 8)
// ---------------------------------------------------------------------------
__global__ void transpose_cvt_kernel(const float* __restrict__ W, bf16_t* __restrict__ Wt,
                                     int K, int N) {
  __shared__ bf16_t tile[32][33];
  int kb = blockIdx.x * 32;
  int nb = blockIdx.y * 32;
  int tx = threadIdx.x;                // 0..31
  int ty = threadIdx.y;                // 0..7
#pragma unroll
  for (int i = 0; i < 32; i += 8)
    tile[ty + i][tx] = (bf16_t)W[(size_t)(kb + ty + i) * N + nb + tx];
  __syncthreads();
#pragma unroll
  for (int i = 0; i < 32; i += 8)
    Wt[(size_t)(nb + ty + i) * K + kb + tx] = tile[tx][ty + i];
}

// ---------------------------------------------------------------------------
// WMMA GEMM, 32x64 output tile per wave (2 M x 4 N subtiles):
//   C(MxN) = A(MxK) @ Bt(NxK)^T + bias
// SHARE_A: all 8 waves of a block share the same 32 A-rows (requires
//          tilesN % 8 == 0); A chunk cooperatively staged in LDS as bf16.
// A_F32: A is fp32 (else bf16). OUT_F32: f32 out (else bf16).
// VTRANS: store bf16 as vt[b][n][s] with S=2048 per batch.
// ---------------------------------------------------------------------------
template <int SHARE_A, int A_F32, int OUT_F32, int VTRANS>
__global__ void __launch_bounds__(256, 1)
gemm32x64(const void* __restrict__ Av, const bf16_t* __restrict__ Bt,
          const float* __restrict__ bias, void* __restrict__ Cv,
          int M, int N, int K, int tilesN) {
  __shared__ bf16_t aT[32][40];        // 32 rows x 32 k, stride 80B (16B-aligned)

  int wave = blockIdx.x * 8 + (threadIdx.x >> 5);
  int tm, tn, m0;
  if constexpr (SHARE_A) {
    tm = (blockIdx.x * 8) / tilesN;    // identical for all waves in block
    tn = wave - tm * tilesN;
    m0 = tm * 32;
  } else {
    tm = wave / tilesN;
    tn = wave - tm * tilesN;
    m0 = tm * 32;
    if (m0 >= M) return;
  }
  int n0 = tn * 64;
  int lane = threadIdx.x & 31;

  v8f c[2][4];
#pragma unroll
  for (int ms = 0; ms < 2; ++ms)
#pragma unroll
    for (int ct = 0; ct < 4; ++ct)
#pragma unroll
      for (int r = 0; r < 8; ++r) c[ms][ct][r] = 0.f;

  int t  = threadIdx.x;
  int mr = t >> 3;                     // 0..31
  int kc = (t & 7) << 2;               // 0..28 step 4

  for (int k0 = 0; k0 < K; k0 += 32) {
    v16bf a0, a1;
    if constexpr (SHARE_A) {
      __syncthreads();                 // prior-iteration readers done
      if constexpr (A_F32) {
        const float* ap = (const float*)Av + (size_t)(m0 + mr) * K + k0 + kc;
        float4 f = *(const float4*)ap;
        aT[mr][kc + 0] = (bf16_t)f.x;
        aT[mr][kc + 1] = (bf16_t)f.y;
        aT[mr][kc + 2] = (bf16_t)f.z;
        aT[mr][kc + 3] = (bf16_t)f.w;
      } else {
        const bf16_t* ap = (const bf16_t*)Av + (size_t)(m0 + mr) * K + k0 + kc;
#pragma unroll
        for (int i = 0; i < 4; ++i) aT[mr][kc + i] = ap[i];
      }
      __syncthreads();
      a0 = load_a_bf16(&aT[0][0], 40, 0, 0);
      a1 = load_a_bf16(&aT[0][0], 40, 16, 0);
    } else {
      if constexpr (A_F32) {
        a0 = load_a_f32((const float*)Av, K, m0, k0);
        a1 = load_a_f32((const float*)Av, K, m0 + 16, k0);
      } else {
        a0 = load_a_bf16((const bf16_t*)Av, K, m0, k0);
        a1 = load_a_bf16((const bf16_t*)Av, K, m0 + 16, k0);
      }
    }
#pragma unroll
    for (int ct = 0; ct < 4; ++ct) {
      v16bf b = load_b_nk(Bt, K, n0 + ct * 16, k0);
      c[0][ct] = wmma_bf16(a0, b, c[0][ct]);
      c[1][ct] = wmma_bf16(a1, b, c[1][ct]);
    }
  }

#pragma unroll
  for (int ms = 0; ms < 2; ++ms) {
    int mbase = m0 + ms * 16 + ((lane >> 4) << 3);
#pragma unroll
    for (int ct = 0; ct < 4; ++ct) {
      int n = n0 + ct * 16 + (lane & 15);
      float bv = bias[n];
#pragma unroll
      for (int r = 0; r < 8; ++r) {
        float val = c[ms][ct][r] + bv;
        int m = mbase + r;
        if constexpr (VTRANS) {
          int bb = m >> 11, msl = m & 2047;                     // S = 2048 per batch
          ((bf16_t*)Cv)[((size_t)bb * N + n) * 2048 + msl] = (bf16_t)val;
        } else if constexpr (OUT_F32) {
          ((float*)Cv)[(size_t)m * N + n] = val;
        } else {
          ((bf16_t*)Cv)[(size_t)m * N + n] = (bf16_t)val;
        }
      }
    }
  }
}

// ---------------------------------------------------------------------------
// Flash-style MQA attention core.
// One wave per (b, h, 16-row Q tile); 8 waves/block share the same batch, so
// each 64-key K/V chunk is cooperatively staged in LDS once per block.
//   qh   : (B*S, 1024) bf16    kh : (B, S, 64) bf16    vt : (B, 64, S) bf16
//   attn : (B*S, 1024) bf16
// ---------------------------------------------------------------------------
__global__ void __launch_bounds__(256, 1)
flash_mqa_kernel(const bf16_t* __restrict__ qh,
                 const bf16_t* __restrict__ kh,
                 const bf16_t* __restrict__ vt,
                 bf16_t* __restrict__ attn) {
  constexpr int S = 2048, D = 1024, HD = 64;
  constexpr int LDP = 72;                    // padded row stride (bank-spread)
  __shared__ bf16_t sK[64 * LDP];            // [kpos_local][hd]
  __shared__ bf16_t sV[64 * LDP];            // [hd][kpos_local]
  __shared__ bf16_t sP[8][16 * 64];          // per-wave P staging

  int wslot = threadIdx.x >> 5;
  int wave  = blockIdx.x * 8 + wslot;
  int qt = wave & 127;                        // S/16 q-tiles
  int h  = (wave >> 7) & 15;
  int b  = wave >> 11;
  int lane = threadIdx.x & 31;
  int q0 = qt * 16;

  const bf16_t* khb = kh + (size_t)b * S * HD;
  const bf16_t* vtb = vt + (size_t)b * HD * S;
  bf16_t* myP = &sP[wslot][0];

  // cooperative-load mapping: 256 threads x 32B cover one 64x64 bf16 chunk
  int crow = threadIdx.x >> 2;               // 0..63
  int ccol = (threadIdx.x & 3) << 4;         // 0,16,32,48

  v16bf aq0 = load_a_bf16(qh, D, b * S + q0, h * HD + 0);
  v16bf aq1 = load_a_bf16(qh, D, b * S + q0, h * HD + 32);

  float mrow[8], lrow[8];
  v8f o[4];
#pragma unroll
  for (int r = 0; r < 8; ++r) { mrow[r] = -1e30f; lrow[r] = 0.f; }
#pragma unroll
  for (int ct = 0; ct < 4; ++ct)
#pragma unroll
    for (int r = 0; r < 8; ++r) o[ct][r] = 0.f;

  for (int k0 = 0; k0 < S; k0 += 64) {
    // ---- stage K chunk (rows = kpos) and V chunk (rows = hd) into LDS
    __syncthreads();
    {
      const bf16_t* gk = khb + (size_t)(k0 + crow) * HD + ccol;
      const bf16_t* gv = vtb + (size_t)crow * S + k0 + ccol;
      v8bf k0v = *(const v8bf*)gk;
      v8bf k1v = *(const v8bf*)(gk + 8);
      v8bf v0v = *(const v8bf*)gv;
      v8bf v1v = *(const v8bf*)(gv + 8);
      *(v8bf*)&sK[crow * LDP + ccol]     = k0v;
      *(v8bf*)&sK[crow * LDP + ccol + 8] = k1v;
      *(v8bf*)&sV[crow * LDP + ccol]     = v0v;
      *(v8bf*)&sV[crow * LDP + ccol + 8] = v1v;
    }
    __syncthreads();

    // ---- S tile: 16 q-rows x 64 keys, contraction over HD=64
    v8f s[4];
#pragma unroll
    for (int ct = 0; ct < 4; ++ct) {
      v16bf bk0 = load_b_lds(sK, LDP, ct * 16, 0);
      v16bf bk1 = load_b_lds(sK, LDP, ct * 16, 32);
      v8f acc = {0.f, 0.f, 0.f, 0.f, 0.f, 0.f, 0.f, 0.f};
      acc = wmma_bf16(aq0, bk0, acc);
      acc = wmma_bf16(aq1, bk1, acc);
#pragma unroll
      for (int r = 0; r < 8; ++r) {
        float x = acc[r] * 0.125f;                      // 1/sqrt(64)
        s[ct][r] = fminf(fmaxf(x, -1e6f), 1e6f);        // clamp per reference
      }
    }

    // ---- online softmax (row reductions across 16 lanes of each half)
    float mnew[8], alpha[8], rsum[8];
#pragma unroll
    for (int r = 0; r < 8; ++r) {
      float mx = fmaxf(fmaxf(s[0][r], s[1][r]), fmaxf(s[2][r], s[3][r]));
#pragma unroll
      for (int off = 1; off < 16; off <<= 1)
        mx = fmaxf(mx, __shfl_xor(mx, off, 16));
      mnew[r]  = fmaxf(mrow[r], mx);
      alpha[r] = __expf(mrow[r] - mnew[r]);
      mrow[r]  = mnew[r];
      rsum[r]  = 0.f;
    }

    // ---- P = exp(s - m), stage to per-wave LDS (C layout -> row-major)
#pragma unroll
    for (int ct = 0; ct < 4; ++ct) {
#pragma unroll
      for (int r = 0; r < 8; ++r) {
        float p = __expf(s[ct][r] - mnew[r]);
        rsum[r] += p;
        int row = ((lane >> 4) << 3) + r;
        myP[row * 64 + ct * 16 + (lane & 15)] = (bf16_t)p;
      }
    }
#pragma unroll
    for (int r = 0; r < 8; ++r) {
      float sm = rsum[r];
#pragma unroll
      for (int off = 1; off < 16; off <<= 1) sm += __shfl_xor(sm, off, 16);
      lrow[r] = lrow[r] * alpha[r] + sm;
    }
#pragma unroll
    for (int ct = 0; ct < 4; ++ct)
#pragma unroll
      for (int r = 0; r < 8; ++r) o[ct][r] *= alpha[r];

    // ---- read P back as A fragments; O += P @ V
    v16bf ap0 = load_a_bf16(myP, 64, 0, 0);
    v16bf ap1 = load_a_bf16(myP, 64, 0, 32);
#pragma unroll
    for (int ct = 0; ct < 4; ++ct) {
      v16bf bv0 = load_b_lds(sV, LDP, ct * 16, 0);
      v16bf bv1 = load_b_lds(sV, LDP, ct * 16, 32);
      o[ct] = wmma_bf16(ap0, bv0, o[ct]);
      o[ct] = wmma_bf16(ap1, bv1, o[ct]);
    }
  }

  // ---- normalize and write per-head output block (bf16)
#pragma unroll
  for (int r = 0; r < 8; ++r) lrow[r] = 1.f / lrow[r];
  int mbase = (lane >> 4) << 3;
#pragma unroll
  for (int ct = 0; ct < 4; ++ct) {
    int col = h * HD + ct * 16 + (lane & 15);
#pragma unroll
    for (int r = 0; r < 8; ++r) {
      int row = b * S + q0 + mbase + r;
      attn[(size_t)row * D + col] = (bf16_t)(o[ct][r] * lrow[r]);
    }
  }
}

// ---------------------------------------------------------------------------
// Host launcher
// ---------------------------------------------------------------------------
extern "C" void kernel_launch(void* const* d_in, const int* in_sizes, int n_in,
                              void* d_out, int out_size, void* d_ws, size_t ws_size,
                              hipStream_t stream) {
  (void)in_sizes; (void)n_in; (void)out_size; (void)ws_size;
  const float* q  = (const float*)d_in[0];
  const float* k  = (const float*)d_in[1];
  const float* v  = (const float*)d_in[2];
  const float* Wq = (const float*)d_in[3];
  const float* bq = (const float*)d_in[4];
  const float* Wk = (const float*)d_in[5];
  const float* bk = (const float*)d_in[6];
  const float* Wv = (const float*)d_in[7];
  const float* bv = (const float*)d_in[8];
  const float* Wo = (const float*)d_in[9];
  const float* bo = (const float*)d_in[10];
  float* out = (float*)d_out;

  // Workspace layout (all 256B-aligned)
  char* ws = (char*)d_ws;
  bf16_t* qh   = (bf16_t*)(ws);                                      // 4096x1024 = 8 MB
  bf16_t* attn = (bf16_t*)(ws + (size_t)(8u  << 20));                // 4096x1024 = 8 MB
  bf16_t* kh   = (bf16_t*)(ws + (size_t)(16u << 20));                // 2x2048x64 = 512 KB
  bf16_t* vt   = (bf16_t*)(ws + (size_t)(16u << 20) + (512u << 10)); // 2x64x2048
  bf16_t* wqt  = (bf16_t*)(ws + (size_t)(17u << 20));                // 1024x1024 = 2 MB
  bf16_t* wot  = (bf16_t*)(ws + (size_t)(19u << 20));                // 1024x1024 = 2 MB
  bf16_t* wkt  = (bf16_t*)(ws + (size_t)(21u << 20));                // 64x1024 = 128 KB
  bf16_t* wvt  = (bf16_t*)(ws + (size_t)(21u << 20) + (128u << 10));

  // 1) weights -> bf16, transposed to N x K
  transpose_cvt_kernel<<<dim3(32, 32), dim3(32, 8), 0, stream>>>(Wq, wqt, 1024, 1024);
  transpose_cvt_kernel<<<dim3(32, 2),  dim3(32, 8), 0, stream>>>(Wk, wkt, 1024, 64);
  transpose_cvt_kernel<<<dim3(32, 2),  dim3(32, 8), 0, stream>>>(Wv, wvt, 1024, 64);
  transpose_cvt_kernel<<<dim3(32, 32), dim3(32, 8), 0, stream>>>(Wo, wot, 1024, 1024);

  // 2) projections (M = 4096 = B*S); 32x64 tile/wave, 8 waves/block
  //    Q: 4096x1024x1024 f32->bf16, A shared via LDS (tilesN=16)
  gemm32x64<1, 1, 0, 0><<<2048 / 8, 256, 0, stream>>>(q, wqt, bq, qh, 4096, 1024, 1024, 16);
  //    K: 4096x64x1024, bf16 out (row-major), tilesN=1 -> direct path
  gemm32x64<0, 1, 0, 0><<<128 / 8, 256, 0, stream>>>(k, wkt, bk, kh, 4096, 64, 1024, 1);
  //    V: 4096x64x1024, bf16 out stored transposed per batch vt[b][64][2048]
  gemm32x64<0, 1, 0, 1><<<128 / 8, 256, 0, stream>>>(v, wvt, bv, vt, 4096, 64, 1024, 1);

  // 3) attention: 2*16*128 = 4096 waves / 8 per block
  flash_mqa_kernel<<<512, 256, 0, stream>>>(qh, kh, vt, attn);

  // 4) output projection: 4096x1024x1024, bf16 A (shared), f32 out
  gemm32x64<1, 0, 1, 0><<<2048 / 8, 256, 0, stream>>>(attn, wot, bo, out, 4096, 1024, 1024, 16);
}
